// KSae_48112223650361
// MI455X (gfx1250) — matrix operand order repaired and checked by reading, not compiled, not run on hardware
//
#include <hip/hip_runtime.h>

// ---------------- problem constants ----------------
#define BATCH   2048
#define D_IN    1024
#define N_LAT   32768
#define TOPK    32
#define K_AUX   512

// ---------------- raw vector types (avoid HIP class-type vector quirks) ----
typedef __attribute__((ext_vector_type(4)))  float   f4x;
typedef __attribute__((ext_vector_type(4)))  int     i4x;
typedef __attribute__((ext_vector_type(16))) __bf16  v16bf;
typedef __attribute__((ext_vector_type(2)))  __bf16  v2bf;
typedef __attribute__((ext_vector_type(8)))  float   v8f;

union Frag { i4x q[2]; v16bf v; };

// round-to-nearest-even f32 -> bf16 (manual fallback)
__device__ __forceinline__ unsigned short f2bf(float f) {
    unsigned u = __float_as_uint(f);
    unsigned r = (u + 0x7FFFu + ((u >> 16) & 1u)) >> 16;
    return (unsigned short)r;
}
__device__ __forceinline__ unsigned pack2(float a, float b) {
#if __has_builtin(__builtin_amdgcn_cvt_pk_bf16_f32)
    v2bf r = __builtin_amdgcn_cvt_pk_bf16_f32(a, b);
    return __builtin_bit_cast(unsigned, r);
#else
    return (unsigned)f2bf(a) | ((unsigned)f2bf(b) << 16);
#endif
}

// ================= K0: zero counts + ws scalars =================
__global__ void sae_init_kernel(float* counts, float* ws) {
    int i = blockIdx.x * 256 + threadIdx.x;
    if (i < N_LAT) counts[i] = 0.0f;
    if (i < 4) ws[i] = 0.0f;
}

// ================= K1: column mean of x =================
__global__ void sae_colmean_kernel(const float* __restrict__ x, float* __restrict__ mean) {
    int col = blockIdx.x * 256 + threadIdx.x;
    if (col >= D_IN) return;
    float s = 0.0f;
    for (int r = 0; r < BATCH; ++r) s += x[(size_t)r * D_IN + col];
    mean[col] = s * (1.0f / (float)BATCH);
}

// ================= K2: total variance =================
__global__ void sae_tv_kernel(const float* __restrict__ x, const float* __restrict__ mean,
                              float* ws) {
    __shared__ float red[256];
    int row = blockIdx.x, tid = threadIdx.x;
    float sq = 0.0f;
#pragma unroll
    for (int c = 0; c < 4; ++c) {
        int col = tid + 256 * c;
        float d = x[(size_t)row * D_IN + col] - mean[col];
        sq += d * d;
    }
    red[tid] = sq; __syncthreads();
    for (int s = 128; s > 0; s >>= 1) {
        if (tid < s) red[tid] += red[tid + s];
        __syncthreads();
    }
    if (tid == 0) atomicAdd(&ws[0], red[0]);
}

// ================= K3: encoder GEMM (bf16 WMMA, f32 accumulate) =================
// pre = relu((x - b_dec) @ W_enc^T + b_enc), tiles BM=128 x BN=128 x BK=64,
// software-pipelined: next chunk's global loads issue while WMMAs consume LDS.
#define BK     64
#define LDSS   72   // padded bf16 stride per row (144B, 16B-aligned)

__global__ __launch_bounds__(256)
void sae_encode_kernel(const float* __restrict__ x, const float* __restrict__ W_enc,
                       const float* __restrict__ b_enc, const float* __restrict__ b_dec,
                       float* __restrict__ pre) {
    __shared__ __align__(16) unsigned short sA[128 * LDSS];
    __shared__ __align__(16) unsigned short sB[128 * LDSS];

    const int tid = threadIdx.x;
    const int blockN = blockIdx.x * 128;   // latent dim
    const int blockM = blockIdx.y * 128;   // batch dim

    const int lane = tid & 31;
    const int wv   = tid >> 5;
    const int mg   = wv & 1;    // 2 m-groups of 64
    const int ng   = wv >> 1;   // 4 n-groups of 32
    const int mr   = lane & 15;
    const int hlf  = lane >> 4;

    v8f acc[4][2];
#pragma unroll
    for (int a = 0; a < 4; ++a)
#pragma unroll
        for (int b = 0; b < 2; ++b)
#pragma unroll
            for (int r = 0; r < 8; ++r) acc[a][b][r] = 0.0f;

    const int krow = tid >> 4;          // 0..15
    const int kq4  = (tid & 15) << 2;   // 0,4,...,60

    // ---- prologue: load chunk 0 into registers ----
    f4x ra[8], rb[8];
#pragma unroll
    for (int p = 0; p < 8; ++p) {
        int rowi = krow + p * 16;
        ra[p] = *(const f4x*)(x     + (size_t)(blockM + rowi) * D_IN + kq4);
        rb[p] = *(const f4x*)(W_enc + (size_t)(blockN + rowi) * D_IN + kq4);
    }

    for (int kc = 0; kc < D_IN; kc += BK) {
        // ---- convert + store current chunk registers to LDS as bf16 ----
        f4x bd = *(const f4x*)(b_dec + kc + kq4);
#pragma unroll
        for (int p = 0; p < 8; ++p) {
            int rowi = krow + p * 16;
            unsigned* dA = (unsigned*)&sA[rowi * LDSS + kq4];
            dA[0] = pack2(ra[p].x - bd.x, ra[p].y - bd.y);
            dA[1] = pack2(ra[p].z - bd.z, ra[p].w - bd.w);
            unsigned* dB = (unsigned*)&sB[rowi * LDSS + kq4];
            dB[0] = pack2(rb[p].x, rb[p].y);
            dB[1] = pack2(rb[p].z, rb[p].w);
        }
        __syncthreads();

        // ---- issue next chunk's global loads; they complete behind the WMMAs ----
        if (kc + BK < D_IN) {
            int kn = kc + BK;
#pragma unroll
            for (int p = 0; p < 8; ++p) {
                int rowi = krow + p * 16;
                ra[p] = *(const f4x*)(x     + (size_t)(blockM + rowi) * D_IN + kn + kq4);
                rb[p] = *(const f4x*)(W_enc + (size_t)(blockN + rowi) * D_IN + kn + kq4);
            }
        }

        // ---- compute: 2 k-steps of K=32 ----
#pragma unroll
        for (int ks = 0; ks < BK; ks += 32) {
            Frag afr[4];
#pragma unroll
            for (int a = 0; a < 4; ++a) {
                const unsigned short* pA = sA + (mg * 64 + a * 16 + mr) * LDSS + ks;
                afr[a].q[0] = *(const i4x*)(pA + 8 * hlf);        // K 0..7  (+8*half)
                afr[a].q[1] = *(const i4x*)(pA + 16 + 8 * hlf);   // K 16..23(+8*half)
            }
            Frag bfr[2];
#pragma unroll
            for (int b = 0; b < 2; ++b) {
                const unsigned short* pB = sB + (ng * 32 + b * 16 + mr) * LDSS + ks;
                bfr[b].q[0] = *(const i4x*)(pB + 8 * hlf);        // K 0..7  (+8*half)
                bfr[b].q[1] = *(const i4x*)(pB + 16 + 8 * hlf);   // K 16..23(+8*half)
            }
#pragma unroll
            for (int a = 0; a < 4; ++a)
#pragma unroll
                for (int b = 0; b < 2; ++b)
                    acc[a][b] = __builtin_amdgcn_wmma_f32_16x16x32_bf16(
                        false, afr[a].v, false, bfr[b].v,
                        (short)0, acc[a][b], false, false);
        }
        __syncthreads();
    }

    // ---- bias + relu + store ----
#pragma unroll
    for (int a = 0; a < 4; ++a) {
#pragma unroll
        for (int b = 0; b < 2; ++b) {
            int gn = blockN + ng * 32 + b * 16 + mr;
            float be = b_enc[gn];
#pragma unroll
            for (int r = 0; r < 8; ++r) {
                int gm = blockM + mg * 64 + a * 16 + r + 8 * hlf;
                float v = fmaxf(acc[a][b][r] + be, 0.0f);
                pre[(size_t)gm * N_LAT + gn] = v;
            }
        }
    }
}

// ================= K4: per-row top-32 (row cached in 128KB LDS) =================
__global__ void sae_topk_kernel(const float* __restrict__ pre, float* __restrict__ top_acts,
                                float* __restrict__ top_idx, float* counts) {
    extern __shared__ float smT[];
    float* vals = smT;                       // 32768
    float* redv = smT + N_LAT;               // 256
    int*   redi = (int*)(smT + N_LAT + 256); // 256

    int row = blockIdx.x, tid = threadIdx.x;
    const float* src = pre + (size_t)row * N_LAT;
    for (int i = tid; i < N_LAT; i += 256) vals[i] = src[i];
    __syncthreads();

    for (int it = 0; it < TOPK; ++it) {
        float bv = -__builtin_inff(); int bi = N_LAT;
        for (int i = tid; i < N_LAT; i += 256) {
            float v = vals[i];
            if (v > bv) { bv = v; bi = i; }
        }
        redv[tid] = bv; redi[tid] = bi;
        __syncthreads();
        for (int s = 128; s > 0; s >>= 1) {
            if (tid < s) {
                float ov = redv[tid + s]; int oi = redi[tid + s];
                if (ov > redv[tid] || (ov == redv[tid] && oi < redi[tid])) {
                    redv[tid] = ov; redi[tid] = oi;
                }
            }
            __syncthreads();
        }
        if (tid == 0) {
            int w = redi[0];
            top_acts[row * TOPK + it] = redv[0];
            top_idx [row * TOPK + it] = (float)w;
            vals[w] = -__builtin_inff();
            atomicAdd(&counts[w], 1.0f);
        }
        __syncthreads();
    }
}

// ================= K5: sparse decode + per-row ||e||^2 =================
__global__ void sae_decode_kernel(const float* __restrict__ x, const float* __restrict__ W_dec,
                                  const float* __restrict__ b_dec,
                                  const float* __restrict__ top_acts,
                                  const float* __restrict__ top_idx,
                                  float* __restrict__ sae_out, float* ws) {
    __shared__ float sAct[TOPK];
    __shared__ int   sIdx[TOPK];
    __shared__ float red[256];
    int row = blockIdx.x, tid = threadIdx.x;
    if (tid < TOPK) {
        sAct[tid] = top_acts[row * TOPK + tid];
        sIdx[tid] = (int)top_idx[row * TOPK + tid];
    }
    __syncthreads();
    float acc[4] = {0.f, 0.f, 0.f, 0.f};
    for (int j = 0; j < TOPK; ++j) {
        const float* wr = W_dec + (size_t)sIdx[j] * D_IN;
        float a = sAct[j];
#pragma unroll
        for (int c = 0; c < 4; ++c) acc[c] += a * wr[tid + 256 * c];
    }
    float sq = 0.0f;
#pragma unroll
    for (int c = 0; c < 4; ++c) {
        int col = tid + 256 * c;
        float o = acc[c] + b_dec[col];
        sae_out[(size_t)row * D_IN + col] = o;
        float e = o - x[(size_t)row * D_IN + col];
        sq += e * e;
    }
    red[tid] = sq; __syncthreads();
    for (int s = 128; s > 0; s >>= 1) {
        if (tid < s) red[tid] += red[tid + s];
        __syncthreads();
    }
    if (tid == 0) atomicAdd(&ws[1], red[0]);
}

// ================= K6: dead mask + num_dead =================
__global__ void sae_dead_kernel(const float* __restrict__ counts, float* __restrict__ deadf,
                                float* ws) {
    __shared__ float red[256];
    int tid = threadIdx.x;
    int i = blockIdx.x * 256 + tid;
    float d = 0.0f;
    if (i < N_LAT) {
        d = (counts[i] == 0.0f) ? 1.0f : 0.0f;
        deadf[i] = d;
    }
    red[tid] = d; __syncthreads();
    for (int s = 128; s > 0; s >>= 1) {
        if (tid < s) red[tid] += red[tid + s];
        __syncthreads();
    }
    if (tid == 0) atomicAdd(&ws[3], red[0]);
}

// ================= K7: auxk radix-select top-512 among dead + loss =================
__global__ void sae_auxk_kernel(const float* __restrict__ pre, const float* __restrict__ deadf,
                                const float* __restrict__ x, const float* __restrict__ sae_out,
                                const float* __restrict__ W_dec, const float* __restrict__ b_dec,
                                float* ws) {
    extern __shared__ float smA[];
    float*    vals   = smA;                                  // 32768
    unsigned* hist   = (unsigned*)(smA + N_LAT);             // 256
    float*    selAct = smA + N_LAT + 256;                    // 512
    int*      selIdx = (int*)(smA + N_LAT + 256 + 512);      // 512
    int*      ctrl   = (int*)(smA + N_LAT + 256 + 1024);     // 4
    float*    red    = smA + N_LAT + 256 + 1024 + 4;         // 256

    int row = blockIdx.x, tid = threadIdx.x;
    const float* src = pre + (size_t)row * N_LAT;
    for (int i = tid; i < N_LAT; i += 256)
        vals[i] = (deadf[i] > 0.5f) ? src[i] : -1.0f;   // relu'd acts are >= 0
    int numDead = (int)(ws[3] + 0.5f);
    if (tid == 0) { ctrl[2] = 0; ctrl[3] = 0; }
    __syncthreads();

    unsigned prefix = 0; int remaining = K_AUX;
    if (numDead > K_AUX) {
        // radix select, MSB digit first; values >= 0 so uint bits are monotone
        for (int d = 3; d >= 0; --d) {
            if (tid < 256) hist[tid] = 0;
            __syncthreads();
            for (int i = tid; i < N_LAT; i += 256) {
                float v = vals[i];
                if (v >= 0.0f) {
                    unsigned b = __float_as_uint(v);
                    unsigned hi = (d == 3) ? 0u : (b >> (8 * (d + 1)));
                    if (hi == prefix) atomicAdd(&hist[(b >> (8 * d)) & 255u], 1u);
                }
            }
            __syncthreads();
            if (tid == 0) {
                int cum = 0, dig = 0, rem2 = remaining;
                for (int bb = 255; bb >= 0; --bb) {
                    int h = (int)hist[bb];
                    if (cum + h >= remaining) { dig = bb; rem2 = remaining - cum; break; }
                    cum += h;
                }
                ctrl[0] = (int)((prefix << 8) | (unsigned)dig);
                ctrl[1] = rem2;
            }
            __syncthreads();
            prefix = (unsigned)ctrl[0];
            remaining = ctrl[1];
        }
    }
    // select: > threshold always, == threshold first `remaining`
    for (int i = tid; i < N_LAT; i += 256) {
        float v = vals[i];
        if (v >= 0.0f) {
            bool take;
            if (numDead <= K_AUX) take = true;
            else {
                unsigned b = __float_as_uint(v);
                if (b > prefix) take = true;
                else if (b == prefix) { int p = atomicAdd(&ctrl[3], 1); take = (p < remaining); }
                else take = false;
            }
            if (take) {
                int s = atomicAdd(&ctrl[2], 1);
                if (s < K_AUX) { selIdx[s] = i; selAct[s] = v; }
            }
        }
    }
    __syncthreads();
    int ns = ctrl[2]; if (ns > K_AUX) ns = K_AUX;

    // e_hat = decode(selected) ; accumulate sum((e_hat - e)^2)
    float acc[4] = {0.f, 0.f, 0.f, 0.f};
    for (int j = 0; j < ns; ++j) {
        const float* wr = W_dec + (size_t)selIdx[j] * D_IN;
        float a = selAct[j];
#pragma unroll
        for (int c = 0; c < 4; ++c) acc[c] += a * wr[tid + 256 * c];
    }
    float sq = 0.0f;
#pragma unroll
    for (int c = 0; c < 4; ++c) {
        int col = tid + 256 * c;
        float eh = acc[c] + b_dec[col];
        float e  = sae_out[(size_t)row * D_IN + col] - x[(size_t)row * D_IN + col];
        float dd = eh - e;
        sq += dd * dd;
    }
    red[tid] = sq; __syncthreads();
    for (int s = 128; s > 0; s >>= 1) {
        if (tid < s) red[tid] += red[tid + s];
        __syncthreads();
    }
    if (tid == 0) atomicAdd(&ws[2], red[0]);
}

// ================= K8: finalize scalars =================
__global__ void sae_finalize_kernel(const float* ws, float* fvu_out, float* auxk_out) {
    if (blockIdx.x == 0 && threadIdx.x == 0) {
        float tv = ws[0];
        float l2 = ws[1] * (1.0f / (float)BATCH);
        *fvu_out  = l2 / tv;
        *auxk_out = (ws[3] > 0.0f) ? (ws[2] / tv) : 0.0f;
    }
}

// ================= host: launch =================
extern "C" void kernel_launch(void* const* d_in, const int* in_sizes, int n_in,
                              void* d_out, int out_size, void* d_ws, size_t ws_size,
                              hipStream_t stream) {
    (void)in_sizes; (void)n_in; (void)out_size; (void)ws_size;
    const float* x     = (const float*)d_in[0];
    // d_in[1] dead_mask: ignored (recomputed by forward)
    const float* W_enc = (const float*)d_in[2];
    const float* b_enc = (const float*)d_in[3];
    const float* W_dec = (const float*)d_in[4];
    const float* b_dec = (const float*)d_in[5];

    float* out      = (float*)d_out;
    float* sae_out  = out;                                   // 2048*1024
    float* pre_acts = sae_out + (size_t)BATCH * D_IN;        // 2048*32768
    float* top_acts = pre_acts + (size_t)BATCH * N_LAT;      // 2048*32
    float* top_idx  = top_acts + (size_t)BATCH * TOPK;       // 2048*32
    float* fvu      = top_idx + (size_t)BATCH * TOPK;        // 1
    float* counts   = fvu + 1;                               // 32768
    float* auxk     = counts + N_LAT;                        // 1

    float* ws     = (float*)d_ws;   // [0]=tv [1]=l2sum [2]=auxk_sq [3]=num_dead
    float* wsmean = ws + 4;         // 1024
    float* deadf  = ws + 4 + D_IN;  // 32768

    sae_init_kernel<<<N_LAT / 256, 256, 0, stream>>>(counts, ws);
    sae_colmean_kernel<<<D_IN / 256, 256, 0, stream>>>(x, wsmean);
    sae_tv_kernel<<<BATCH, 256, 0, stream>>>(x, wsmean, ws);

    dim3 gridE(N_LAT / 128, BATCH / 128);
    sae_encode_kernel<<<gridE, 256, 0, stream>>>(x, W_enc, b_enc, b_dec, pre_acts);

    size_t topkLds = (size_t)(N_LAT + 256) * sizeof(float) + 256 * sizeof(int);
    sae_topk_kernel<<<BATCH, 256, topkLds, stream>>>(pre_acts, top_acts, top_idx, counts);

    sae_decode_kernel<<<BATCH, 256, 0, stream>>>(x, W_dec, b_dec, top_acts, top_idx,
                                                 sae_out, ws);
    sae_dead_kernel<<<N_LAT / 256, 256, 0, stream>>>(counts, deadf, ws);

    size_t auxkLds = (size_t)(N_LAT + 256 + 1024 + 4 + 256) * sizeof(float);
    sae_auxk_kernel<<<BATCH, 256, auxkLds, stream>>>(pre_acts, deadf, x, sae_out,
                                                     W_dec, b_dec, ws);
    sae_finalize_kernel<<<1, 64, 0, stream>>>(ws, fvu, auxk);
}
